// SafaBiasedCrossAttention_5935644803512
// MI455X (gfx1250) — compile-verified, matrix operand order checked by reading, not run on hardware
//
#include <hip/hip_runtime.h>
#include <hip/hip_bf16.h>
#include <cstdint>
#include <cstddef>

#define B_     64
#define C_     256
#define NIMG   1024
#define NEXTRA 32
#define NTOK   1056   // 1024 + 32, = 33 * 32 exactly
#define NH     8
#define DK     64

typedef __attribute__((ext_vector_type(16))) __bf16 v16bf;
typedef __attribute__((ext_vector_type(8)))  __bf16 v8bf;
typedef __attribute__((ext_vector_type(8)))  float  v8f;
typedef __attribute__((ext_vector_type(4)))  unsigned u32x4;
typedef __attribute__((ext_vector_type(8)))  unsigned u32x8;

// ---------- helpers ----------
__device__ __forceinline__ unsigned short f2bf_bits(float f) {
  unsigned u = __float_as_uint(f);
  unsigned r = u + 0x7FFFu + ((u >> 16) & 1u);   // round-to-nearest-even
  return (unsigned short)(r >> 16);
}
__device__ __forceinline__ unsigned pack2bf(float a, float b) {
  return (unsigned)f2bf_bits(a) | ((unsigned)f2bf_bits(b) << 16);
}
__device__ __forceinline__ float sigmoidf_(float z) {
  return 1.0f / (1.0f + __expf(-z));
}
// Load a 16x32 bf16 WMMA fragment whose K runs contiguously in memory.
// p points at (row, K = k0 + half*8); elements 8..15 live at K offset +16.
__device__ __forceinline__ v16bf load_frag_k(const unsigned short* p) {
  v8bf lo = *(const v8bf*)(p);
  v8bf hi = *(const v8bf*)(p + 16);
  return __builtin_shufflevector(lo, hi, 0,1,2,3,4,5,6,7,8,9,10,11,12,13,14,15);
}
// Async-DMA 32 bytes/lane of a table into LDS (ASYNCcnt path).
__device__ __forceinline__ void async_copy32(void* ldsp, const void* gp) {
  unsigned ldsa = (unsigned)(uintptr_t)ldsp;
  unsigned long long ga = (unsigned long long)(uintptr_t)gp;
  asm volatile("global_load_async_to_lds_b128 %0, %1, off" :: "v"(ldsa), "v"(ga) : "memory");
  asm volatile("global_load_async_to_lds_b128 %0, %1, off offset:16" :: "v"(ldsa + 16u), "v"(ga) : "memory");
}
// Issue one TDM 2D-tile load (32 rows x 128B, row stride 4KB) into LDS.
// g1 carries the static tile geometry; g0 is built here (count=1, type=image).
__device__ __forceinline__ void tdm_tile_load(unsigned ldsa, unsigned long long ga, u32x8 g1) {
  u32x4 g0;
  g0[0] = 1u;                                              // count = 1 valid D#
  g0[1] = ldsa;                                            // lds_addr
  g0[2] = (unsigned)ga;                                    // global_addr[31:0]
  g0[3] = ((unsigned)(ga >> 32) & 0x01ffffffu) | 0x80000000u; // addr[56:32] | type=2
  asm volatile("tensor_load_to_lds %0, %1" :: "s"(g0), "s"(g1) : "memory");
}

// ---------- workspace layout (bytes) ----------
#define OFF_V    0ull                      // [8][256] f32            8 KB
#define OFF_P    8192ull                   // [8][1024] f32           32 KB
#define OFF_CM   40960ull                  // [64][1024] bf16         128 KB
#define OFF_X    172032ull                 // [64][8][1056] f32       2.06 MB
#define OFF_ATTN 2334720ull                // [64][8][1056] bf16      1.03 MB
#define OFF_TOK  3416064ull                // [64][256][1056] bf16    33 MB (c-major!)

// ============================================================
// Kernel 1: fold cls into W_k and pos_embed (with (1-gate)/sqrt(dk))
// ============================================================
__global__ void precompute_kernel(const float* __restrict__ Wk,
                                  const float* __restrict__ cls,
                                  const float* __restrict__ pos,
                                  const float* __restrict__ gatep,
                                  float* __restrict__ v,
                                  float* __restrict__ p) {
  int idx = blockIdx.x * 256 + threadIdx.x;
  float s = (1.0f - sigmoidf_(gatep[0])) * 0.125f;
  if (idx < NH * C_) {
    int h = idx / C_, c = idx % C_;
    float acc = 0.f;
    for (int d = 0; d < DK; ++d)
      acc = fmaf(Wk[(h * C_ + c) * DK + d], cls[h * DK + d], acc);
    v[idx] = acc * s;
  } else if (idx < NH * C_ + NH * NIMG) {
    int j = idx - NH * C_;
    int h = j / NIMG, n = j % NIMG;
    float acc = 0.f;
    for (int d = 0; d < DK; ++d)
      acc = fmaf(pos[n * DK + d], cls[h * DK + d], acc);
    p[j] = acc * s;
  }
}

// ============================================================
// Kernel 2a: single fp32 pass over features (4-wide vectorized).
// Produces: channel-max (bf16), c-major bf16 tokens, folded xattn logits.
// v-table staged to LDS via global_load_async_to_lds_b128.
// ============================================================
__global__ __launch_bounds__(256) void pass1_img_kernel(
    const float* __restrict__ feat, const float* __restrict__ v,
    const float* __restrict__ p, float* __restrict__ x,
    unsigned short* __restrict__ cm, unsigned short* __restrict__ tokT) {
  __shared__ __align__(16) float vlds[NH * C_];
  const int tid = threadIdx.x;
  async_copy32(&vlds[tid * 8], v + tid * 8);               // 8 KB table -> LDS
  asm volatile("s_wait_asynccnt 0" ::: "memory");
  __syncthreads();

  const int b = blockIdx.x;
  const int n0 = tid * 4;                                  // 4 pixels per lane
  float m[4];
  float acc[NH][4];
#pragma unroll
  for (int i = 0; i < 4; ++i) m[i] = -3.4e38f;
#pragma unroll
  for (int h = 0; h < NH; ++h)
#pragma unroll
    for (int i = 0; i < 4; ++i) acc[h][i] = 0.f;

  for (int c = 0; c < C_; ++c) {
    const float* rowp = feat + (size_t)(b * C_ + c) * NIMG + n0;
    float4 f = *(const float4*)rowp;                       // b128, coalesced
    if (c + 1 < C_) __builtin_prefetch(rowp + NIMG, 0, 1);
    float fv[4] = {f.x, f.y, f.z, f.w};
#pragma unroll
    for (int i = 0; i < 4; ++i) m[i] = fmaxf(m[i], fv[i]);
#pragma unroll
    for (int h = 0; h < NH; ++h) {
      float wv_ = vlds[h * C_ + c];                        // LDS broadcast
#pragma unroll
      for (int i = 0; i < 4; ++i) acc[h][i] = fmaf(fv[i], wv_, acc[h][i]);
    }
    uint2 tk; tk.x = pack2bf(fv[0], fv[1]); tk.y = pack2bf(fv[2], fv[3]);
    *(uint2*)(tokT + (size_t)(b * C_ + c) * NTOK + n0) = tk;   // b64, coalesced
  }
  uint2 cmv; cmv.x = pack2bf(m[0], m[1]); cmv.y = pack2bf(m[2], m[3]);
  *(uint2*)(cm + b * NIMG + n0) = cmv;
#pragma unroll
  for (int h = 0; h < NH; ++h) {
    const float4 pv = *(const float4*)(p + h * NIMG + n0);
    float4 xs;
    xs.x = acc[h][0] + pv.x; xs.y = acc[h][1] + pv.y;
    xs.z = acc[h][2] + pv.z; xs.w = acc[h][3] + pv.w;
    *(float4*)(x + (size_t)(b * NH + h) * NTOK + n0) = xs;     // b128 store
  }
}

// ============================================================
// Kernel 2b: extra tokens (2 MB, latency-trivial)
// ============================================================
__global__ void pass1_extra_kernel(const float* __restrict__ extra,
                                   const float* __restrict__ v,
                                   float* __restrict__ x,
                                   unsigned short* __restrict__ tokT) {
  __shared__ __align__(16) float vlds[NH * C_];
  const int tid = threadIdx.x;
  async_copy32(&vlds[tid * 8], v + tid * 8);
  asm volatile("s_wait_asynccnt 0" ::: "memory");
  __syncthreads();
  int g = blockIdx.x * 256 + tid;        // 0..2047 => (b, e)
  int b = g / NEXTRA, e = g % NEXTRA;
  float acc[NH];
#pragma unroll
  for (int h = 0; h < NH; ++h) acc[h] = 0.f;
  for (int c = 0; c < C_; ++c) {
    float val = extra[(size_t)(b * NEXTRA + e) * C_ + c];
#pragma unroll
    for (int h = 0; h < NH; ++h) acc[h] = fmaf(val, vlds[h * C_ + c], acc[h]);
    tokT[(size_t)(b * C_ + c) * NTOK + NIMG + e] = f2bf_bits(val);
  }
#pragma unroll
  for (int h = 0; h < NH; ++h)
    x[(size_t)(b * NH + h) * NTOK + NIMG + e] = acc[h];  // no pos embed here
}

// ============================================================
// Kernel 3: SAFA GEMM + fused gated-sigmoid attention.
//   S[b, j] = sum_i cm_bf16[b,i] * bf16(safa_W[h,i,j]);  M=64, N=32/block, K=1024
//   attn = sigmoid(gate*(S + safa_b) + x)  -> bf16
// safa_W tiles are streamed by the Tensor Data Mover (TENSORcnt) into a
// ping-pong LDS buffer: wave0 issues tile ks+1 before consuming tile ks,
// so the DMA overlaps convert+WMMA. f32->bf16 pack lands directly in the
// ISA B-fragment swizzle (two K-adjacent values per ds_store_b32).
// ============================================================
__global__ __launch_bounds__(256) void safa_attn_kernel(
    const float* __restrict__ safaW, const float* __restrict__ safab,
    const float* __restrict__ gatep, const unsigned short* __restrict__ cm,
    const float* __restrict__ x, unsigned short* __restrict__ attn) {
  __shared__ __align__(16) float rawW[2][32 * 32];           // 8 KB ping-pong fp32 tiles
  __shared__ __align__(16) unsigned short fragB[2][32][16];  // 2 swizzled B frags

  const int tid  = threadIdx.x;
  const int lane = tid & 31;
  const int wv   = tid >> 5;
  const int mt   = wv & 3;   // which 16-row block of b (M=64)
  const int nh   = wv >> 2;  // which 16-col half of the 32-wide j tile
  const int h    = blockIdx.x;
  const int j0   = blockIdx.y * 32;

  const int halfA = lane >> 4;
  const int mrow  = mt * 16 + (lane & 15);
  const unsigned short* arow = cm + mrow * NIMG;

  // Static TDM descriptor group1: data_size=4B, tensor_dim0=1024,
  // tensor_dim1=1024, tile 32x32, dim0 stride 1024 elements.
  const u32x8 g1 = {2u << 16, 1024u << 16, 1024u << 16, 32u << 16,
                    32u, 1024u, 0u, 0u};
  const unsigned lb0 = (unsigned)(uintptr_t)&rawW[0][0];
  const unsigned lb1 = (unsigned)(uintptr_t)&rawW[1][0];
  const unsigned long long gb = (unsigned long long)(uintptr_t)(
      (const char*)safaW + 4ull * ((size_t)(h * NIMG) * NIMG + j0));

  if (wv == 0) tdm_tile_load(lb0, gb, g1);                 // prologue: tile 0

  v8f acc = {};
  for (int ks = 0; ks < 32; ++ks) {
    const int cur = ks & 1;
    if (wv == 0) {
      if (ks < 31) {   // stream next tile while this one is consumed
        tdm_tile_load(cur ? lb0 : lb1, gb + 0x20000ull * (unsigned long long)(ks + 1), g1);
        __builtin_amdgcn_s_wait_tensorcnt(1);              // tile ks landed
      } else {
        __builtin_amdgcn_s_wait_tensorcnt(0);
      }
    }
    if (wv == 1 && ks < 30)                                // L2 prefetch, 2 tiles ahead
      __builtin_prefetch((const void*)(uintptr_t)(
          gb + 0x20000ull * (unsigned long long)(ks + 2) + (unsigned long long)lane * 4096ull), 0, 1);
    __syncthreads();   // tile ks visible; fragB free (prev wmma reads done)

    // f32 -> bf16 + swizzle into the ISA B-fragment layout.
    // element e=2g+r of lane (N%16 + 16*half) holds K=(g>=4)*16+half*8+(g%4)*2+r;
    // K and K+1 (K even) share a VGPR -> pack pairs, store b32.
#pragma unroll
    for (int pp = 0; pp < 2; ++pp) {
      int idx = pp * 256 + tid;                 // 0..511
      int kk  = (idx >> 5) * 2;                 // even K
      int nn  = idx & 31;
      int e   = ((kk >= 16) ? 8 : 0) + (kk & 7);
      int hf  = (kk & 15) >> 3;
      unsigned pk = pack2bf(rawW[cur][kk * 32 + nn], rawW[cur][(kk + 1) * 32 + nn]);
      *(unsigned*)&fragB[nn >> 4][(nn & 15) + 16 * hf][e] = pk;
    }
    __syncthreads();

    // A fragment straight from global (K contiguous in cm rows)
    const int k0 = ks * 32;
    v16bf a = load_frag_k(arow + k0 + halfA * 8);
    // B fragment from LDS (two ds_load_b128 per lane)
    const unsigned short* bp = &fragB[nh][lane][0];
    v8bf blo = *(const v8bf*)(bp);
    v8bf bhi = *(const v8bf*)(bp + 8);
    v16bf bb = __builtin_shufflevector(blo, bhi, 0,1,2,3,4,5,6,7,8,9,10,11,12,13,14,15);

    acc = __builtin_amdgcn_wmma_f32_16x16x32_bf16(false, a, false, bb,
                                                  (short)0, acc, false, false);
  }

  // D layout: vgpr g -> (M = g + 8*(lane>=16), N = lane%16)
  const float gate = sigmoidf_(gatep[0]);
  const int j  = j0 + nh * 16 + (lane & 15);
  const int hf = lane >> 4;
  const float bias = safab[h * NIMG + j];
#pragma unroll
  for (int g = 0; g < 8; ++g) {
    int brow = mt * 16 + g + 8 * hf;
    float logits = acc[g] + bias;
    float xv = x[(size_t)(brow * NH + h) * NTOK + j];
    attn[(size_t)(brow * NH + h) * NTOK + j] =
        f2bf_bits(sigmoidf_(gate * logits + xv));
  }
}

// ============================================================
// Kernel 3b: attention for the 32 extra tokens (safa pad = 0)
// ============================================================
__global__ void attn_extra_kernel(const float* __restrict__ x,
                                  unsigned short* __restrict__ attn) {
  int b = blockIdx.x, t = threadIdx.x;   // 256 = 8 heads * 32 extras
  int h = t >> 5, e = t & 31;
  size_t i = (size_t)(b * NH + h) * NTOK + NIMG + e;
  attn[i] = f2bf_bits(sigmoidf_(x[i]));
}

// ============================================================
// Kernel 4: out[b,h,c] = sum_n attn[b,h,n] * tokens[b,n,c]
// A = attn [M=16 (h, top 8 rows zero), K=1056], B = tokT [K=n, N=c].
// Both operands have K contiguous => pure-register WMMA, no LDS.
// ============================================================
__global__ __launch_bounds__(256) void out_gemm_kernel(
    const unsigned short* __restrict__ attn,
    const unsigned short* __restrict__ tokT, float* __restrict__ out) {
  const int tid = threadIdx.x, lane = tid & 31, wv = tid >> 5;
  const int b = blockIdx.x;
  const int c0 = blockIdx.y * 128 + wv * 16;
  const int halfK = lane >> 4;
  const int hrow  = lane & 15;                  // A-matrix M for this lane
  const int cidx  = c0 + (lane & 15);           // B-matrix N for this lane

  const unsigned short* browp = tokT + (size_t)(b * C_ + cidx) * NTOK;
  const unsigned short* arowp = attn + (size_t)(b * NH + (hrow & 7)) * NTOK;
  const bool avalid = hrow < NH;

  v8f acc = {};
  for (int ks = 0; ks < 33; ++ks) {             // K = 1056 = 33*32
    const int k0 = ks * 32;
    v8bf alo = {}, ahi = {};
    if (avalid) {
      const unsigned short* ap = arowp + k0 + halfK * 8;
      alo = *(const v8bf*)(ap);
      ahi = *(const v8bf*)(ap + 16);
    }
    v16bf a  = __builtin_shufflevector(alo, ahi, 0,1,2,3,4,5,6,7,8,9,10,11,12,13,14,15);
    v16bf bb = load_frag_k(browp + k0 + halfK * 8);
    if (ks < 32) __builtin_prefetch(browp + k0 + 32, 0, 1);
    acc = __builtin_amdgcn_wmma_f32_16x16x32_bf16(false, a, false, bb,
                                                  (short)0, acc, false, false);
  }
  // D: lanes 0..15 hold M=g (valid heads 0..7); lanes 16..31 hold M=g+8 (pad)
  if (lane < 16) {
#pragma unroll
    for (int g = 0; g < 8; ++g)
      out[(size_t)b * (NH * C_) + g * C_ + (c0 + lane)] = acc[g];
  }
}

// ============================================================
extern "C" void kernel_launch(void* const* d_in, const int* in_sizes, int n_in,
                              void* d_out, int out_size, void* d_ws, size_t ws_size,
                              hipStream_t stream) {
  (void)in_sizes; (void)n_in; (void)out_size; (void)ws_size;
  const float* features = (const float*)d_in[0];
  const float* extra    = (const float*)d_in[1];
  const float* safaW    = (const float*)d_in[2];
  const float* safab    = (const float*)d_in[3];
  const float* cls      = (const float*)d_in[4];
  const float* Wk       = (const float*)d_in[5];
  const float* pos      = (const float*)d_in[6];
  const float* gatep    = (const float*)d_in[7];
  float* out = (float*)d_out;

  char* ws = (char*)d_ws;
  float* v            = (float*)(ws + OFF_V);
  float* p            = (float*)(ws + OFF_P);
  unsigned short* cm  = (unsigned short*)(ws + OFF_CM);
  float* x            = (float*)(ws + OFF_X);
  unsigned short* at  = (unsigned short*)(ws + OFF_ATTN);
  unsigned short* tok = (unsigned short*)(ws + OFF_TOK);

  precompute_kernel<<<40, 256, 0, stream>>>(Wk, cls, pos, gatep, v, p);
  pass1_img_kernel<<<B_, 256, 0, stream>>>(features, v, p, x, cm, tok);
  pass1_extra_kernel<<<8, 256, 0, stream>>>(extra, v, x, tok);
  safa_attn_kernel<<<dim3(NH, 32), 256, 0, stream>>>(safaW, safab, gatep, cm, x, at);
  attn_extra_kernel<<<B_, 256, 0, stream>>>(x, at);
  out_gemm_kernel<<<dim3(B_, 2), 256, 0, stream>>>(at, tok, out);
}
